// _Attention_21646635172126
// MI455X (gfx1250) — matrix operand order repaired and checked by reading, not compiled, not run on hardware
//
#include <hip/hip_runtime.h>

// ---------------------------------------------------------------------------
// Problem constants
// ---------------------------------------------------------------------------
#define DIMS   768
#define HEADS  12
#define HD     64
#define NSEQ   197
#define NPAD   224              // 14 tiles of 16, 7 k-steps of 32
#define BATCH  64
#define MROWS  (BATCH * NSEQ)   // 12608 = 788 * 16 = 197 * 64
#define QSCALE 0.125f           // 64^-0.5

typedef __attribute__((ext_vector_type(16))) __bf16        v16bf;
typedef __attribute__((ext_vector_type(8)))  float         v8f;
typedef __attribute__((ext_vector_type(8)))  unsigned int  v8u;
typedef __attribute__((ext_vector_type(4)))  unsigned int  v4u;

static __device__ __forceinline__ __bf16 f2bf(float f) {
  unsigned u = __builtin_bit_cast(unsigned, f);
  u += 0x7fffu + ((u >> 16) & 1u);              // round-to-nearest-even
  unsigned short h = (unsigned short)(u >> 16);
  return __builtin_bit_cast(__bf16, h);
}

static __device__ __forceinline__ v8f vzero8() {
  v8f z = {0.f, 0.f, 0.f, 0.f, 0.f, 0.f, 0.f, 0.f};
  return z;
}

// ---------------------------------------------------------------------------
// WMMA bf16 fragment loaders (wave32), per CDNA5 ISA 7.12.2 layouts.
//
// A operand: 16x32 (MxK) bf16, source row-major with row stride ld.
//   lane: m = lane&15, half = lane>>4
//   vgpr v packs K = (v>>2)*16 + half*8 + (v&3)*2 + {0,1}
//   -> per lane: two contiguous 16-byte runs  => 2x b128 loads.
static __device__ __forceinline__ v16bf frag_a(const __bf16* base, int ld, int lane) {
  int m = lane & 15, half = (lane >> 4) & 1;
  const __bf16* p = base + (size_t)m * ld + half * 8;
  v4u lo = *(const v4u*)(p);
  v4u hi = *(const v4u*)(p + 16);
  v8u f = __builtin_shufflevector(lo, hi, 0, 1, 2, 3, 4, 5, 6, 7);
  return __builtin_bit_cast(v16bf, f);
}

// B operand: 32x16 (KxN) bf16; element(K, n) = src[n*ld + K] (i.e. we pass the
// row-major *transpose* of B, which is how k / Wᵀ / Vᵀ are stored).
//   lane: n = lane&15, half = lane>>4
//   vgpr v packs K = (v>>2)*16 + half*8 + (v&3)*2 + {0,1}  -> two 16-byte runs.
static __device__ __forceinline__ v16bf frag_b(const __bf16* base, int ld, int lane) {
  int n = lane & 15, half = (lane >> 4) & 1;
  const __bf16* p = base + (size_t)n * ld + half * 8;
  v4u lo = *(const v4u*)(p);
  v4u hi = *(const v4u*)(p + 16);
  v8u f = __builtin_shufflevector(lo, hi, 0, 1, 2, 3, 4, 5, 6, 7);
  return __builtin_bit_cast(v16bf, f);
}

#define WMMA_BF16(a, b, c) \
  __builtin_amdgcn_wmma_f32_16x16x32_bf16(false, (a), false, (b), (short)0, (c), false, false)

// ---------------------------------------------------------------------------
// Kernel 0: f32 -> bf16 conversion
// ---------------------------------------------------------------------------
__global__ __launch_bounds__(256) void k_cvt_bf16(const float* __restrict__ in,
                                                  __bf16* __restrict__ out, int n) {
  int i = blockIdx.x * 256 + threadIdx.x;
  if (i < n) out[i] = f2bf(in[i]);
}

// ---------------------------------------------------------------------------
// Kernel 1: qkv = x @ qkv_wᵀ + bias  -> scattered into Q (scaled), K, Vᵀ (bf16)
// One wave computes a 64x64 C tile. grid = (197, 36), block = 32.
// ---------------------------------------------------------------------------
__global__ __launch_bounds__(32) void k_gemm_qkv(
    const __bf16* __restrict__ X, const __bf16* __restrict__ W,
    const float* __restrict__ qb, const float* __restrict__ vb,
    __bf16* __restrict__ Q, __bf16* __restrict__ K, __bf16* __restrict__ VT) {
  int lane = threadIdx.x;
  int tm = blockIdx.x, tn = blockIdx.y;

  v8f acc[4][4];
#pragma unroll
  for (int i = 0; i < 4; ++i)
#pragma unroll
    for (int j = 0; j < 4; ++j) acc[i][j] = vzero8();

  const __bf16* xa = X + (size_t)tm * 64 * DIMS;
  const __bf16* wb = W + (size_t)tn * 64 * DIMS;

  for (int kt = 0; kt < DIMS; kt += 32) {
    v16bf a[4], b[4];
#pragma unroll
    for (int i = 0; i < 4; ++i) a[i] = frag_a(xa + (size_t)i * 16 * DIMS + kt, DIMS, lane);
#pragma unroll
    for (int j = 0; j < 4; ++j) b[j] = frag_b(wb + (size_t)j * 16 * DIMS + kt, DIMS, lane);
    if (kt + 32 < DIMS) {                       // gfx1250 global_prefetch_b8
      __builtin_prefetch(xa + (size_t)(lane & 3) * 16 * DIMS + kt + 32, 0, 1);
      __builtin_prefetch(wb + (size_t)(lane & 3) * 16 * DIMS + kt + 32, 0, 1);
    }
#pragma unroll
    for (int i = 0; i < 4; ++i)
#pragma unroll
      for (int j = 0; j < 4; ++j) acc[i][j] = WMMA_BF16(a[i], b[j], acc[i][j]);
  }

  // C layout: vgpr r -> row r + 8*(lane>>4); col = lane&15
  int nl = lane & 15, mh = (lane >> 4) << 3;
#pragma unroll
  for (int j = 0; j < 4; ++j) {
    int gc = tn * 64 + j * 16 + nl;             // 0..2303
    int s = gc / DIMS;                          // 0=q 1=k 2=v
    int rem = gc - s * DIMS;
    int h = rem >> 6, d = rem & 63;
    float bias = (s == 0) ? qb[rem] : ((s == 2) ? vb[rem] : 0.f);
#pragma unroll
    for (int i = 0; i < 4; ++i) {
#pragma unroll
      for (int r = 0; r < 8; ++r) {
        int gm = tm * 64 + i * 16 + r + mh;     // < 12608 always
        int b_ = gm / NSEQ;
        int nn = gm - b_ * NSEQ;
        float val = acc[i][j][r] + bias;
        if (s == 0) val *= QSCALE;
        size_t bh = (size_t)b_ * HEADS + h;
        if (s == 0)      Q[(bh * NPAD + nn) * HD + d] = f2bf(val);
        else if (s == 1) K[(bh * NPAD + nn) * HD + d] = f2bf(val);
        else             VT[(bh * HD + d) * NPAD + nn] = f2bf(val);
      }
    }
  }
}

// ---------------------------------------------------------------------------
// Kernel 2: attention for one (b, h, 16-row tile). block = 32 (one wave).
// grid = (13, HEADS, BATCH).
// ---------------------------------------------------------------------------
__global__ __launch_bounds__(32) void k_attn(
    const __bf16* __restrict__ Q, const __bf16* __restrict__ K,
    const __bf16* __restrict__ VT, const float* __restrict__ rel_table,
    const int* __restrict__ rel_index, __bf16* __restrict__ CTX) {
  __shared__ alignas(16) __bf16 P[16 * NPAD];

  int lane = threadIdx.x;
  int it = blockIdx.x, h = blockIdx.y, b = blockIdx.z;
  int rowbase = it * 16;
  size_t bh = (size_t)b * HEADS + h;
  const __bf16* qp = Q + bh * NPAD * HD;
  const __bf16* kp = K + bh * NPAD * HD;
  const __bf16* vt = VT + bh * HD * NPAD;

  int nl = lane & 15, mh = (lane >> 4) << 3;

  // S = (q*scale) kᵀ  + rel_bias; 16 rows x 224 cols kept in registers
  v16bf a0 = frag_a(qp + (size_t)rowbase * HD, HD, lane);
  v16bf a1 = frag_a(qp + (size_t)rowbase * HD + 32, HD, lane);

  v8f s[14];
#pragma unroll
  for (int j = 0; j < 14; ++j) {
    v16bf b0 = frag_b(kp + (size_t)j * 16 * HD, HD, lane);
    v16bf b1 = frag_b(kp + (size_t)j * 16 * HD + 32, HD, lane);
    v8f c = vzero8();
    c = WMMA_BF16(a0, b0, c);
    c = WMMA_BF16(a1, b1, c);
    int jc = j * 16 + nl;
#pragma unroll
    for (int r = 0; r < 8; ++r) {
      int i = rowbase + r + mh;
      float vv = c[r];
      if (jc >= NSEQ)      vv = -1e30f;                              // mask pad cols
      else if (i < NSEQ)   vv += rel_table[rel_index[i * NSEQ + jc] * HEADS + h];
      c[r] = vv;
    }
    s[j] = c;
  }

  // softmax: each row lives in one vgpr slot across one 16-lane half
#pragma unroll
  for (int r = 0; r < 8; ++r) {
    float mx = -1e30f;
#pragma unroll
    for (int j = 0; j < 14; ++j) mx = fmaxf(mx, s[j][r]);
    for (int d = 1; d < 16; d <<= 1) mx = fmaxf(mx, __shfl_xor(mx, d, 32));
    float sum = 0.f;
#pragma unroll
    for (int j = 0; j < 14; ++j) {
      float e = __expf(s[j][r] - mx);
      s[j][r] = e;
      sum += e;
    }
    for (int d = 1; d < 16; d <<= 1) sum += __shfl_xor(sum, d, 32);
    float inv = 1.f / sum;
#pragma unroll
    for (int j = 0; j < 14; ++j) s[j][r] *= inv;
  }

  // reshape C-layout -> A-layout through LDS (bf16)
#pragma unroll
  for (int j = 0; j < 14; ++j)
#pragma unroll
    for (int r = 0; r < 8; ++r)
      P[(r + mh) * NPAD + j * 16 + nl] = f2bf(s[j][r]);
  __syncthreads();

  // out = P @ V   (P: 16x224, Vᵀ stored as [d][n] so frag_b is contiguous-K)
  v8f o[4];
#pragma unroll
  for (int dt = 0; dt < 4; ++dt) o[dt] = vzero8();
#pragma unroll
  for (int kt = 0; kt < 7; ++kt) {
    v16bf pa = frag_a(P + kt * 32, NPAD, lane);
#pragma unroll
    for (int dt = 0; dt < 4; ++dt) {
      v16bf pb = frag_b(vt + (size_t)dt * 16 * NPAD + kt * 32, NPAD, lane);
      o[dt] = WMMA_BF16(pa, pb, o[dt]);
    }
  }

  // ctx[b*197+i][h*64+d] bf16
#pragma unroll
  for (int dt = 0; dt < 4; ++dt)
#pragma unroll
    for (int r = 0; r < 8; ++r) {
      int i = rowbase + r + mh;
      if (i < NSEQ) {
        size_t gm = (size_t)b * NSEQ + i;
        int col = h * HD + dt * 16 + nl;
        CTX[gm * DIMS + col] = f2bf(o[dt][r]);
      }
    }
}

// ---------------------------------------------------------------------------
// Kernel 3: out = ctx @ proj_wᵀ + proj_b (f32 out). grid = (197, 12), block 32.
// ---------------------------------------------------------------------------
__global__ __launch_bounds__(32) void k_gemm_proj(
    const __bf16* __restrict__ CTX, const __bf16* __restrict__ W,
    const float* __restrict__ pb, float* __restrict__ out) {
  int lane = threadIdx.x;
  int tm = blockIdx.x, tn = blockIdx.y;

  v8f acc[4][4];
#pragma unroll
  for (int i = 0; i < 4; ++i)
#pragma unroll
    for (int j = 0; j < 4; ++j) acc[i][j] = vzero8();

  const __bf16* xa = CTX + (size_t)tm * 64 * DIMS;
  const __bf16* wb = W + (size_t)tn * 64 * DIMS;

  for (int kt = 0; kt < DIMS; kt += 32) {
    v16bf a[4], b[4];
#pragma unroll
    for (int i = 0; i < 4; ++i) a[i] = frag_a(xa + (size_t)i * 16 * DIMS + kt, DIMS, lane);
#pragma unroll
    for (int j = 0; j < 4; ++j) b[j] = frag_b(wb + (size_t)j * 16 * DIMS + kt, DIMS, lane);
    if (kt + 32 < DIMS) {
      __builtin_prefetch(xa + (size_t)(lane & 3) * 16 * DIMS + kt + 32, 0, 1);
      __builtin_prefetch(wb + (size_t)(lane & 3) * 16 * DIMS + kt + 32, 0, 1);
    }
#pragma unroll
    for (int i = 0; i < 4; ++i)
#pragma unroll
      for (int j = 0; j < 4; ++j) acc[i][j] = WMMA_BF16(a[i], b[j], acc[i][j]);
  }

  int nl = lane & 15, mh = (lane >> 4) << 3;
#pragma unroll
  for (int j = 0; j < 4; ++j) {
    int gc = tn * 64 + j * 16 + nl;
    float bias = pb[gc];
#pragma unroll
    for (int i = 0; i < 4; ++i)
#pragma unroll
      for (int r = 0; r < 8; ++r) {
        int gm = tm * 64 + i * 16 + r + mh;
        out[(size_t)gm * DIMS + gc] = acc[i][j][r] + bias;
      }
  }
}

// ---------------------------------------------------------------------------
// Launch
// ---------------------------------------------------------------------------
extern "C" void kernel_launch(void* const* d_in, const int* in_sizes, int n_in,
                              void* d_out, int out_size, void* d_ws, size_t ws_size,
                              hipStream_t stream) {
  (void)in_sizes; (void)n_in; (void)out_size; (void)ws_size;

  const float* x      = (const float*)d_in[0];
  const float* qkv_w  = (const float*)d_in[1];
  const float* q_bias = (const float*)d_in[2];
  const float* v_bias = (const float*)d_in[3];
  const float* rel_t  = (const float*)d_in[4];
  const int*   rel_i  = (const int*)d_in[5];
  const float* proj_w = (const float*)d_in[6];
  const float* proj_b = (const float*)d_in[7];
  float* out = (float*)d_out;

  // workspace carve-up (~105 MB total)
  char* ws = (char*)d_ws;
  size_t off = 0;
  auto take = [&](size_t bytes) -> void* {
    void* p = ws + off;
    off += (bytes + 255) & ~(size_t)255;
    return p;
  };
  __bf16* XBF   = (__bf16*)take((size_t)MROWS * DIMS * 2);
  __bf16* WQKV  = (__bf16*)take((size_t)3 * DIMS * DIMS * 2);
  __bf16* WPROJ = (__bf16*)take((size_t)DIMS * DIMS * 2);
  __bf16* Qb    = (__bf16*)take((size_t)BATCH * HEADS * NPAD * HD * 2);
  __bf16* Kb    = (__bf16*)take((size_t)BATCH * HEADS * NPAD * HD * 2);
  __bf16* VTb   = (__bf16*)take((size_t)BATCH * HEADS * HD * NPAD * 2);
  __bf16* CTX   = (__bf16*)take((size_t)MROWS * DIMS * 2);

  int nx = MROWS * DIMS;
  k_cvt_bf16<<<dim3((nx + 255) / 256), 256, 0, stream>>>(x, XBF, nx);
  int nw1 = 3 * DIMS * DIMS;
  k_cvt_bf16<<<dim3((nw1 + 255) / 256), 256, 0, stream>>>(qkv_w, WQKV, nw1);
  int nw2 = DIMS * DIMS;
  k_cvt_bf16<<<dim3((nw2 + 255) / 256), 256, 0, stream>>>(proj_w, WPROJ, nw2);

  k_gemm_qkv<<<dim3(MROWS / 64, (3 * DIMS) / 64), 32, 0, stream>>>(
      XBF, WQKV, q_bias, v_bias, Qb, Kb, VTb);

  k_attn<<<dim3((NSEQ + 15) / 16, HEADS, BATCH), 32, 0, stream>>>(
      Qb, Kb, VTb, rel_t, rel_i, CTX);

  k_gemm_proj<<<dim3(MROWS / 64, DIMS / 64), 32, 0, stream>>>(CTX, WPROJ, proj_b, out);
}
